// DecoderLSTMAttn_86260123174493
// MI455X (gfx1250) — matrix operand order repaired and checked by reading, not compiled, not run on hardware
//
#include <hip/hip_runtime.h>

// ---------------------------------------------------------------------------
// Problem constants (from the reference)
// ---------------------------------------------------------------------------
#define BB 256      // batch
#define LL 24       // caption length
#define RR 49       // regions
#define FF 2048     // feature dim
#define EE 256      // embed dim
#define HH 512      // hidden
#define AA 512      // attention dim
#define VV 10000    // vocab
#define VVPAD 10048 // vocab padded to multiple of 64 (157*64)
#define STEPS (LL - 1)
#define KCAT (EE + FF + HH)   // 2816 : [emb | ctx | h]

typedef __bf16 bf16_t;
typedef bf16_t bf16x16 __attribute__((ext_vector_type(16)));
typedef float  f32x8   __attribute__((ext_vector_type(8)));

#define FLAG_TANH    1
#define FLAG_ROWMASK 2

#define MT 2   // 16x16 M-tiles per wave
#define NT 4   // 16x16 N-tiles per wave

static __device__ __forceinline__ bf16_t f2bf(float f) {
  union { float f; unsigned u; } a; a.f = f;
  unsigned u = a.u;
  unsigned r = u + 0x7FFFu + ((u >> 16) & 1u);   // round-to-nearest-even
  union { unsigned short s; bf16_t b; } o;
  o.s = (unsigned short)(r >> 16);
  return o.b;
}

static __device__ __forceinline__ float sigmoidf_(float x) {
  return 1.0f / (1.0f + __expf(-x));
}

// ---------------------------------------------------------------------------
// bf16 WMMA GEMM with 2x4 register blocking:
//   C(M,N) = act( A(M,K) @ W(Npad,K)^T + bias ),  store only n < N
// Each wave owns a 32x64 macro-tile: per K-step, 2 A-frags + 4 B-frags feed
// 8 v_wmma_f32_16x16x32_bf16 (0.75 KB of loads per WMMA vs 2 KB unblocked).
// Requires M % 32 == 0, Npad % 64 == 0, K % 32 == 0, N % 16 == 0.
// ---------------------------------------------------------------------------
__global__ __launch_bounds__(256) void wmma_gemm_bf16(
    const bf16_t* __restrict__ A, int lda,
    const bf16_t* __restrict__ W, int ldw,
    const float*  __restrict__ bias,
    float*        __restrict__ C, long long ldc,
    bf16_t*       __restrict__ Cbf, int ldcbf,
    int M, int N, int Npad, int K, int flags,
    const int* __restrict__ lengths, int tstep)
{
  const int lane = threadIdx.x & 31;
  const int wave = threadIdx.x >> 5;
  const int mMacs = M >> 5;       // macro tiles along M (32 rows each)
  const int nMacs = Npad >> 6;    // macro tiles along N (64 cols each)
  const int mac = blockIdx.x * 8 + wave;
  if (mac >= mMacs * nMacs) return;     // wave-uniform: EXEC stays all-1s

  const int mM = mac / nMacs;
  const int nM = mac % nMacs;
  const int half = lane >> 4;           // 0: lanes 0-15, 1: lanes 16-31
  const int l15  = lane & 15;

  const bf16_t* aRow[MT];
  const bf16_t* wRow[NT];
#pragma unroll
  for (int i = 0; i < MT; ++i)
    aRow[i] = A + (size_t)(mM * 32 + i * 16 + l15) * (size_t)lda;
#pragma unroll
  for (int j = 0; j < NT; ++j)
    wRow[j] = W + (size_t)(nM * 64 + j * 16 + l15) * (size_t)ldw;

  f32x8 acc[MT][NT];
#pragma unroll
  for (int i = 0; i < MT; ++i)
#pragma unroll
    for (int j = 0; j < NT; ++j)
      acc[i][j] = (f32x8){0.f, 0.f, 0.f, 0.f, 0.f, 0.f, 0.f, 0.f};

  union Frag { bf16x16 v; uint4 q[2]; };

  for (int kt = 0; kt < K; kt += 32) {
    Frag a[MT], b[NT];
#pragma unroll
    for (int i = 0; i < MT; ++i) {
      // A 16x32 frag: lanes<16 -> K kt..kt+7 & kt+16..23 ; lanes>=16 shifted by 8
      const bf16_t* ap = aRow[i] + kt + half * 8;
      a[i].q[0] = *(const uint4*)(ap);
      a[i].q[1] = *(const uint4*)(ap + 16);
    }
#pragma unroll
    for (int j = 0; j < NT; ++j) {
      // B 32x16 frag: lanes<16 -> K kt..kt+7 & kt+16..23 ; lanes>=16 shifted by 8
      const bf16_t* wp = wRow[j] + kt + half * 8;
      b[j].q[0] = *(const uint4*)(wp);
      b[j].q[1] = *(const uint4*)(wp + 16);
    }
#pragma unroll
    for (int i = 0; i < MT; ++i)
#pragma unroll
      for (int j = 0; j < NT; ++j)
        acc[i][j] = __builtin_amdgcn_wmma_f32_16x16x32_bf16(
            false, a[i].v, false, b[j].v, (short)0, acc[i][j], false, false);
  }

  // Epilogue. C/D layout: VGPR v -> M = v (lanes 0-15) / v+8 (lanes 16-31); N = lane&15
#pragma unroll
  for (int j = 0; j < NT; ++j) {
    const int n = nM * 64 + j * 16 + l15;
    if (n >= N) continue;               // only hit in the padded vocab tiles
    const float bn = bias ? bias[n] : 0.0f;
#pragma unroll
    for (int i = 0; i < MT; ++i) {
      const int mBase = mM * 32 + i * 16 + half * 8;
#pragma unroll
      for (int v = 0; v < 8; ++v) {
        const int m = mBase + v;
        float val = acc[i][j][v] + bn;
        if (flags & FLAG_TANH) val = tanhf(val);
        if (flags & FLAG_ROWMASK) { if (!(lengths[m] > tstep)) val = 0.0f; }
        C[(size_t)m * (size_t)ldc + n] = val;
        if (Cbf) Cbf[(size_t)m * (size_t)ldcbf + n] = f2bf(val);
      }
    }
  }
}

// ---------------------------------------------------------------------------
// Prep kernels
// ---------------------------------------------------------------------------
__global__ void cvt_f32_bf16(const float* __restrict__ src,
                             bf16_t* __restrict__ dst, long n) {
  for (long i = (long)blockIdx.x * blockDim.x + threadIdx.x; i < n;
       i += (long)gridDim.x * blockDim.x)
    dst[i] = f2bf(src[i]);
}

__global__ void zero_bf16(bf16_t* __restrict__ p, long n) {
  for (long i = (long)blockIdx.x * blockDim.x + threadIdx.x; i < n;
       i += (long)gridDim.x * blockDim.x)
    p[i] = f2bf(0.0f);
}

// Wcat(4H, E+F+H) = [W_ih | W_hh] concatenated along K, in bf16
__global__ void build_wcat(const float* __restrict__ Wih,
                           const float* __restrict__ Whh,
                           bf16_t* __restrict__ wcat, long total) {
  for (long i = (long)blockIdx.x * blockDim.x + threadIdx.x; i < total;
       i += (long)gridDim.x * blockDim.x) {
    int n = (int)(i / KCAT);
    int k = (int)(i % KCAT);
    float v = (k < EE + FF) ? Wih[(size_t)n * (EE + FF) + k]
                            : Whh[(size_t)n * HH + (k - (EE + FF))];
    wcat[i] = f2bf(v);
  }
}

__global__ void build_bcat(const float* __restrict__ bih,
                           const float* __restrict__ bhh,
                           float* __restrict__ bcat) {
  int j = blockIdx.x * blockDim.x + threadIdx.x;
  if (j < 4 * HH) bcat[j] = bih[j] + bhh[j];
}

// emb_bf[b,l,e] = bf16( embed_W[ ids[b,l], e ] )
__global__ void embed_gather(const float* __restrict__ embW,
                             const int* __restrict__ ids,
                             bf16_t* __restrict__ embbf, long total) {
  for (long i = (long)blockIdx.x * blockDim.x + threadIdx.x; i < total;
       i += (long)gridDim.x * blockDim.x) {
    int  e  = (int)(i % EE);
    long bl = i / EE;                     // b*L + l
    int  id = ids[bl];
    embbf[i] = f2bf(embW[(size_t)id * EE + e]);
  }
}

// mean over regions, stored bf16 (only consumed as WMMA A-matrix)
__global__ __launch_bounds__(256) void mean_kernel(
    const float* __restrict__ feats, bf16_t* __restrict__ meanbf) {
  const int b = blockIdx.x;
  for (int f = threadIdx.x; f < FF; f += 256) {
    float s = 0.0f;
    const float* p = feats + (size_t)b * RR * FF + f;
#pragma unroll 7
    for (int r = 0; r < RR; ++r) s += p[(size_t)r * FF];
    meanbf[(size_t)b * FF + f] = f2bf(s * (1.0f / (float)RR));
  }
}

// ---------------------------------------------------------------------------
// Attention: scores -> softmax -> ctx, and packs xcat = [emb_t | ctx | h]
// One block per batch row, 8 waves. shfl_xor wave reductions + LDS softmax.
// ---------------------------------------------------------------------------
__global__ __launch_bounds__(256) void attention_kernel(
    const float*  __restrict__ fproj,   // (B,R,A)
    const float*  __restrict__ hp,      // (B,A)
    const float*  __restrict__ Ws,      // (A)
    const float*  __restrict__ bs,      // (1)
    const float*  __restrict__ feats,   // (B,R,F)
    const bf16_t* __restrict__ emb,     // (B,L,E)
    const bf16_t* __restrict__ hbf,     // (B,H)
    bf16_t*       __restrict__ xcat,    // (B, KCAT)
    int tstep)
{
  const int b    = blockIdx.x;
  const int tid  = threadIdx.x;
  const int lane = tid & 31;
  const int wave = tid >> 5;

  __shared__ float s_al[64];  // softmax weights
  __shared__ float s_sc[64];  // raw scores

  // scores[r] = sum_a tanh(fproj[b,r,a] + hp[b,a]) * Ws[a] + bs
  for (int r = wave; r < RR; r += 8) {
    const float* fp = fproj + ((size_t)b * RR + r) * AA;
    float partial = 0.0f;
    for (int a = lane; a < AA; a += 32)
      partial += tanhf(fp[a] + hp[(size_t)b * AA + a]) * Ws[a];
    for (int off = 16; off > 0; off >>= 1)
      partial += __shfl_xor(partial, off, 32);
    if (lane == 0) s_sc[r] = partial + bs[0];
  }
  __syncthreads();

  if (tid == 0) {
    float m = -1e30f;
    for (int r = 0; r < RR; ++r) m = fmaxf(m, s_sc[r]);
    float s = 0.0f;
    for (int r = 0; r < RR; ++r) { float e = __expf(s_sc[r] - m); s_al[r] = e; s += e; }
    float inv = 1.0f / s;
    for (int r = 0; r < RR; ++r) s_al[r] *= inv;
  }
  __syncthreads();

  bf16_t* xb = xcat + (size_t)b * KCAT;

  // ctx -> xcat[E .. E+F)
  for (int f = tid; f < FF; f += 256) {
    float acc = 0.0f;
    const float* fb = feats + (size_t)b * RR * FF + f;
#pragma unroll 7
    for (int r = 0; r < RR; ++r) acc += s_al[r] * fb[(size_t)r * FF];
    xb[EE + f] = f2bf(acc);
  }
  // emb_t -> xcat[0 .. E)
  for (int e = tid; e < EE; e += 256)
    xb[e] = emb[((size_t)b * LL + tstep) * EE + e];
  // h -> xcat[E+F .. KCAT)
  for (int i = tid; i < HH; i += 256)
    xb[EE + FF + i] = hbf[(size_t)b * HH + i];
}

// ---------------------------------------------------------------------------
// LSTM pointwise: gates -> (h_t, c_t), ragged masking, bf16 copies for GEMMs.
// logits use h_t (pre-mask); carried h/c only update where lengths > t.
// ---------------------------------------------------------------------------
__global__ __launch_bounds__(256) void lstm_pointwise(
    const float* __restrict__ gates,   // (B, 4H) order i,f,g,o
    float* __restrict__ h, float* __restrict__ c,
    bf16_t* __restrict__ hbf,          // carried h, bf16 (next-step GEMM input)
    bf16_t* __restrict__ htbf,         // h_t, bf16 (logits GEMM input)
    const int* __restrict__ lengths, int tstep)
{
  const int b = blockIdx.x;
  const bool active = lengths[b] > tstep;
  const float* g = gates + (size_t)b * 4 * HH;
  for (int j = threadIdx.x; j < HH; j += 256) {
    const float ig = sigmoidf_(g[j]);
    const float fg = sigmoidf_(g[HH + j]);
    const float gg = tanhf(g[2 * HH + j]);
    const float og = sigmoidf_(g[3 * HH + j]);
    const size_t idx = (size_t)b * HH + j;
    const float ct = fg * c[idx] + ig * gg;
    const float ht = og * tanhf(ct);
    htbf[idx] = f2bf(ht);
    if (active) { h[idx] = ht; c[idx] = ct; }
    hbf[idx] = f2bf(h[idx]);
  }
}

// ---------------------------------------------------------------------------
// Host side
// ---------------------------------------------------------------------------
static inline int gemm_blocks(int M, int Npad) {
  return ((M >> 5) * (Npad >> 6) + 7) / 8;
}

extern "C" void kernel_launch(void* const* d_in, const int* in_sizes, int n_in,
                              void* d_out, int out_size, void* d_ws, size_t ws_size,
                              hipStream_t stream) {
  (void)in_sizes; (void)n_in; (void)out_size; (void)ws_size;

  const float* feats   = (const float*)d_in[0];
  const int*   ids     = (const int*)  d_in[1];
  const int*   lengths = (const int*)  d_in[2];
  const float* embW    = (const float*)d_in[3];
  const float* Wf      = (const float*)d_in[4];
  const float* bf_     = (const float*)d_in[5];
  const float* Wh      = (const float*)d_in[6];
  const float* bh      = (const float*)d_in[7];
  const float* Ws      = (const float*)d_in[8];
  const float* bs      = (const float*)d_in[9];
  const float* Wih     = (const float*)d_in[10];
  const float* bih     = (const float*)d_in[11];
  const float* Whh     = (const float*)d_in[12];
  const float* bhh     = (const float*)d_in[13];
  const float* Wfc     = (const float*)d_in[14];
  const float* bfc     = (const float*)d_in[15];
  const float* Wi_h    = (const float*)d_in[16];
  const float* bi_h    = (const float*)d_in[17];
  const float* Wi_c    = (const float*)d_in[18];
  const float* bi_c    = (const float*)d_in[19];

  // ---- workspace layout (256B aligned) ----
  char* base = (char*)d_ws;
  size_t off = 0;
  auto alloc = [&](size_t bytes) -> void* {
    off = (off + 255) & ~(size_t)255;
    void* r = base + off;
    off += bytes;
    return r;
  };
  bf16_t* featsbf = (bf16_t*)alloc((size_t)BB * RR * FF * 2);
  float*  fproj   = (float*) alloc((size_t)BB * RR * AA * 4);
  bf16_t* Wfbf    = (bf16_t*)alloc((size_t)AA * FF * 2);
  bf16_t* Whbf    = (bf16_t*)alloc((size_t)AA * HH * 2);
  bf16_t* Wfcbf   = (bf16_t*)alloc((size_t)VVPAD * HH * 2);   // padded rows zeroed
  bf16_t* Wcatbf  = (bf16_t*)alloc((size_t)4 * HH * KCAT * 2);
  float*  bcat    = (float*) alloc((size_t)4 * HH * 4);
  bf16_t* Wihbf   = (bf16_t*)alloc((size_t)HH * FF * 2);
  bf16_t* Wicbf   = (bf16_t*)alloc((size_t)HH * FF * 2);
  bf16_t* embbf   = (bf16_t*)alloc((size_t)BB * LL * EE * 2);
  bf16_t* meanbf  = (bf16_t*)alloc((size_t)BB * FF * 2);
  float*  hp      = (float*) alloc((size_t)BB * AA * 4);
  bf16_t* xcatbf  = (bf16_t*)alloc((size_t)BB * KCAT * 2);
  float*  gates   = (float*) alloc((size_t)BB * 4 * HH * 4);
  float*  h       = (float*) alloc((size_t)BB * HH * 4);
  float*  c       = (float*) alloc((size_t)BB * HH * 4);
  bf16_t* hbf     = (bf16_t*)alloc((size_t)BB * HH * 2);
  bf16_t* htbf    = (bf16_t*)alloc((size_t)BB * HH * 2);

  auto cvt = [&](const float* s, bf16_t* d, long n) {
    int blk = (int)((n + 255) / 256); if (blk > 8192) blk = 8192;
    cvt_f32_bf16<<<blk, 256, 0, stream>>>(s, d, n);
  };

  // ---- prep ----
  cvt(Wf,   Wfbf,  (long)AA * FF);
  cvt(Wh,   Whbf,  (long)AA * HH);
  cvt(Wfc,  Wfcbf, (long)VV * HH);
  zero_bf16<<<((VVPAD - VV) * HH + 255) / 256, 256, 0, stream>>>(
      Wfcbf + (size_t)VV * HH, (long)(VVPAD - VV) * HH);
  cvt(Wi_h, Wihbf, (long)HH * FF);
  cvt(Wi_c, Wicbf, (long)HH * FF);
  cvt(feats, featsbf, (long)BB * RR * FF);
  {
    long tot = (long)4 * HH * KCAT;
    build_wcat<<<8192, 256, 0, stream>>>(Wih, Whh, Wcatbf, tot);
    build_bcat<<<(4 * HH + 255) / 256, 256, 0, stream>>>(bih, bhh, bcat);
    long te = (long)BB * LL * EE;
    embed_gather<<<(int)((te + 255) / 256), 256, 0, stream>>>(embW, ids, embbf, te);
    mean_kernel<<<BB, 256, 0, stream>>>(feats, meanbf);
  }

  // ---- init state: h0/c0 = tanh(mean @ Wi^T + bi) ----
  wmma_gemm_bf16<<<gemm_blocks(BB, HH), 256, 0, stream>>>(
      meanbf, FF, Wihbf, FF, bi_h, h, HH, hbf, HH,
      BB, HH, HH, FF, FLAG_TANH, nullptr, 0);
  wmma_gemm_bf16<<<gemm_blocks(BB, HH), 256, 0, stream>>>(
      meanbf, FF, Wicbf, FF, bi_c, c, HH, nullptr, 0,
      BB, HH, HH, FF, FLAG_TANH, nullptr, 0);

  // ---- f_proj = feats @ Wf^T + bf  (hoisted) ----
  wmma_gemm_bf16<<<gemm_blocks(BB * RR, AA), 256, 0, stream>>>(
      featsbf, FF, Wfbf, FF, bf_, fproj, AA, nullptr, 0,
      BB * RR, AA, AA, FF, 0, nullptr, 0);

  // ---- time loop ----
  float* out = (float*)d_out;
  for (int t = 0; t < STEPS; ++t) {
    // hp = h @ Wh^T + bh
    wmma_gemm_bf16<<<gemm_blocks(BB, AA), 256, 0, stream>>>(
        hbf, HH, Whbf, HH, bh, hp, AA, nullptr, 0,
        BB, AA, AA, HH, 0, nullptr, 0);
    // attention + pack xcat = [emb_t | ctx | h]
    attention_kernel<<<BB, 256, 0, stream>>>(
        fproj, hp, Ws, bs, feats, embbf, hbf, xcatbf, t);
    // gates = xcat @ [W_ih|W_hh]^T + (b_ih + b_hh)
    wmma_gemm_bf16<<<gemm_blocks(BB, 4 * HH), 256, 0, stream>>>(
        xcatbf, KCAT, Wcatbf, KCAT, bcat, gates, 4 * HH, nullptr, 0,
        BB, 4 * HH, 4 * HH, KCAT, 0, nullptr, 0);
    // LSTM pointwise + masking
    lstm_pointwise<<<BB, 256, 0, stream>>>(gates, h, c, hbf, htbf, lengths, t);
    // logits = mask ? h_t @ Wfc^T + bfc : 0, written into (B, STEPS, V) slice t
    wmma_gemm_bf16<<<gemm_blocks(BB, VVPAD), 256, 0, stream>>>(
        htbf, HH, Wfcbf, HH, bfc, out + (size_t)t * VV, (long long)STEPS * VV,
        nullptr, 0, BB, VV, VVPAD, HH, FLAG_ROWMASK, lengths, t);
  }
}
